// LSTM_8100308320633
// MI455X (gfx1250) — matrix operand order repaired and checked by reading, not compiled, not run on hardware
//
#include <hip/hip_runtime.h>

#define RNN 128
#define GATES 512
#define OBS_LEN 8
#define PRED_LEN 12
#define ROWS_PER_WG 64
#define HSTRIDE 136   // padded row stride (halves): 272B = 17*16B -> aligned, bank-spread
#define WSTRIDE 136

typedef __attribute__((ext_vector_type(16))) _Float16 v16h;
typedef __attribute__((ext_vector_type(8)))  _Float16 v8h;
typedef __attribute__((ext_vector_type(8)))  float    v8f;

__device__ __forceinline__ float sigmoidf_(float x) { return 1.0f / (1.0f + __expf(-x)); }
__device__ __forceinline__ float tanh_f_(float x)   { return 2.0f / (1.0f + __expf(-2.0f * x)) - 1.0f; }

__global__ __launch_bounds__(256, 1)
void lstm_traj_kernel(const float* __restrict__ traj_in,
                      const float* __restrict__ traj_gt,
                      const float* __restrict__ W_ih,
                      const float* __restrict__ W_hh,
                      const float* __restrict__ b_ih,
                      const float* __restrict__ b_hh,
                      const float* __restrict__ W_last,
                      const float* __restrict__ b_last,
                      float* __restrict__ out_pred,
                      float* __restrict__ ws_partial,
                      int B)
{
    // Resident weights + state in LDS for the whole 19-step recurrence.
    __shared__ _Float16 w_lds[GATES * WSTRIDE];      // W_hh as f16, [col][k], padded
    __shared__ _Float16 h_lds[ROWS_PER_WG * HSTRIDE];// h as f16, [row][k], padded
    __shared__ float    bias_lds[GATES];             // b_ih + b_hh
    __shared__ float    wih_lds[GATES * 2];          // W_ih [col][2]
    __shared__ float    wl_lds[2 * RNN];             // W_last [2][128]
    __shared__ float    bl_lds[2];
    __shared__ float    x_lds[ROWS_PER_WG * 2];      // current 2-dim input per row
    __shared__ float    red[ROWS_PER_WG];

    const int tid     = threadIdx.x;
    const int lane    = tid & 31;
    const int wv      = tid >> 5;           // 8 waves
    const int rowbase = blockIdx.x * ROWS_PER_WG;

    // ---- cooperative preload (f32 -> f16 for matmul operands) ----
    for (int e = tid; e < GATES * RNN; e += 256) {
        int col = e >> 7, k = e & 127;
        w_lds[col * WSTRIDE + k] = (_Float16)W_hh[e];
    }
    for (int c = tid; c < GATES; c += 256) {
        bias_lds[c]     = b_ih[c] + b_hh[c];
        wih_lds[2*c]    = W_ih[2*c];
        wih_lds[2*c+1]  = W_ih[2*c+1];
    }
    for (int e = tid; e < 2 * RNN; e += 256) wl_lds[e] = W_last[e];
    if (tid < 2) bl_lds[tid] = b_last[tid];
    for (int e = tid; e < ROWS_PER_WG * HSTRIDE; e += 256) h_lds[e] = (_Float16)0.0f;
    __syncthreads();

    // Wave 'wv' owns hidden columns [j0, j0+16) and all 4 gates for them.
    const int j0 = wv * 16;
    const int ln = lane & 15;
    const int hi = lane >> 4;   // half-wave select

    float bias_g[4], w0_g[4], w1_g[4];
#pragma unroll
    for (int g = 0; g < 4; ++g) {
        int col = g * RNN + j0 + ln;
        bias_g[g] = bias_lds[col];
        w0_g[g]   = wih_lds[2*col];
        w1_g[g]   = wih_lds[2*col+1];
    }

    v8f cst[4];
#pragma unroll
    for (int m = 0; m < 4; ++m)
#pragma unroll
        for (int i = 0; i < 8; ++i) cst[m][i] = 0.0f;

    float localErr = 0.0f;

    for (int s = 0; s < OBS_LEN + PRED_LEN - 1; ++s) {   // 19 cell evals
        if (s < OBS_LEN && tid < ROWS_PER_WG) {
            int gr = rowbase + tid;
            x_lds[2*tid]   = traj_in[(gr * OBS_LEN + s) * 2 + 0];
            x_lds[2*tid+1] = traj_in[(gr * OBS_LEN + s) * 2 + 1];
        }
        __syncthreads();   // x ready, h ready (from prev step / zero-init)

        // ---- accumulator init: bias + x @ W_ih^T (IN=2, done scalar) ----
        v8f acc[4][4];     // [m_tile][gate]
#pragma unroll
        for (int m = 0; m < 4; ++m) {
#pragma unroll
            for (int i = 0; i < 8; ++i) {
                int rl = m * 16 + hi * 8 + i;                 // C-layout row
                float x0 = x_lds[2*rl], x1 = x_lds[2*rl+1];
#pragma unroll
                for (int g = 0; g < 4; ++g)
                    acc[m][g][i] = fmaf(x1, w1_g[g], fmaf(x0, w0_g[g], bias_g[g]));
            }
        }

        // ---- gate GEMM: gates += h @ W_hh^T via v_wmma_f32_16x16x32_f16 ----
        for (int kk = 0; kk < 4; ++kk) {
            const int k0 = kk * 32;
            v16h afrag[4];
#pragma unroll
            for (int m = 0; m < 4; ++m) {
                const _Float16* pa = &h_lds[(m * 16 + ln) * HSTRIDE + k0 + hi * 8];
                union { v16h v; v8h h[2]; } ua;
                ua.h[0] = *(const v8h*)pa;        // K k0(+8*hi) .. +7
                ua.h[1] = *(const v8h*)(pa + 16); // K +16 .. +23
                afrag[m] = ua.v;
            }
#pragma unroll
            for (int g = 0; g < 4; ++g) {
                const int col = g * RNN + j0 + ln;
                const _Float16* pb = &w_lds[col * WSTRIDE + k0 + hi * 8];
                union { v16h v; v8h h[2]; } ub;
                ub.h[0] = *(const v8h*)pb;        // K k0(+8*hi) .. +7
                ub.h[1] = *(const v8h*)(pb + 16); // K +16 .. +23
                v16h bfrag = ub.v;
#pragma unroll
                for (int m = 0; m < 4; ++m)
                    acc[m][g] = __builtin_amdgcn_wmma_f32_16x16x32_f16(
                        false, afrag[m], false, bfrag, (short)0, acc[m][g], false, false);
            }
        }
        __syncthreads();   // all waves done reading h_lds

        // ---- elementwise LSTM cell; c stays in VGPRs, new h -> LDS (f16) ----
#pragma unroll
        for (int m = 0; m < 4; ++m) {
#pragma unroll
            for (int i = 0; i < 8; ++i) {
                float ig = sigmoidf_(acc[m][0][i]);
                float fg = sigmoidf_(acc[m][1][i]);
                float gg = tanh_f_  (acc[m][2][i]);
                float og = sigmoidf_(acc[m][3][i]);
                float c  = fmaf(fg, cst[m][i], ig * gg);
                cst[m][i] = c;
                float h  = og * tanh_f_(c);
                int rl = m * 16 + hi * 8 + i;
                h_lds[rl * HSTRIDE + j0 + ln] = (_Float16)h;
            }
        }
        __syncthreads();   // new h visible

        // ---- head: relu(h) @ W_last^T + b_last; emit pred, feed next input ----
        if (s >= OBS_LEN - 1 && tid < ROWS_PER_WG) {
            int p  = s - (OBS_LEN - 1);        // 0..11
            int gr = rowbase + tid;
            float s0 = bl_lds[0], s1 = bl_lds[1];
            for (int j = 0; j < RNN; ++j) {
                float hv = fmaxf((float)h_lds[tid * HSTRIDE + j], 0.0f);
                s0 = fmaf(hv, wl_lds[j],       s0);
                s1 = fmaf(hv, wl_lds[RNN + j], s1);
            }
            out_pred[(gr * PRED_LEN + p) * 2 + 0] = s0;
            out_pred[(gr * PRED_LEN + p) * 2 + 1] = s1;
            float d0 = s0 - traj_gt[(gr * PRED_LEN + p) * 2 + 0];
            float d1 = s1 - traj_gt[(gr * PRED_LEN + p) * 2 + 1];
            localErr += d0 * d0 + d1 * d1;
            x_lds[2*tid]   = s0;               // next pred-step input
            x_lds[2*tid+1] = s1;
        }
        // top-of-loop __syncthreads covers the x_lds write
    }

    // ---- deterministic per-WG loss partial ----
    if (tid < ROWS_PER_WG) red[tid] = localErr;
    __syncthreads();
    if (tid == 0) {
        float t = 0.0f;
        for (int r = 0; r < ROWS_PER_WG; ++r) t += red[r];
        ws_partial[blockIdx.x] = t;
    }
}

__global__ void lstm_loss_finalize(const float* __restrict__ ws,
                                   float* __restrict__ loss_out,
                                   int nwg, float inv_n)
{
    if (blockIdx.x == 0 && threadIdx.x == 0) {
        float t = 0.0f;
        for (int i = 0; i < nwg; ++i) t += ws[i];
        loss_out[0] = t * inv_n;
    }
}

extern "C" void kernel_launch(void* const* d_in, const int* in_sizes, int n_in,
                              void* d_out, int out_size, void* d_ws, size_t ws_size,
                              hipStream_t stream) {
    const float* traj_in = (const float*)d_in[0];
    const float* traj_gt = (const float*)d_in[1];
    const float* W_ih    = (const float*)d_in[2];
    const float* W_hh    = (const float*)d_in[3];
    const float* b_ih    = (const float*)d_in[4];
    const float* b_hh    = (const float*)d_in[5];
    const float* W_last  = (const float*)d_in[6];
    const float* b_last  = (const float*)d_in[7];

    const int B   = in_sizes[0] / (OBS_LEN * 2);   // 32768
    const int nwg = B / ROWS_PER_WG;               // 512

    float* out = (float*)d_out;
    float* ws  = (float*)d_ws;

    lstm_traj_kernel<<<nwg, 256, 0, stream>>>(traj_in, traj_gt, W_ih, W_hh, b_ih, b_hh,
                                              W_last, b_last, out, ws, B);

    const float inv_n = 1.0f / (float)(B * PRED_LEN * 2);
    lstm_loss_finalize<<<1, 32, 0, stream>>>(ws, out + (size_t)B * PRED_LEN * 2, nwg, inv_n);
}
